// SelfAttention_3590592660316
// MI455X (gfx1250) — compile-verified, matrix operand order checked by reading, not run on hardware
//
#include <hip/hip_runtime.h>
#include <hip/hip_bf16.h>

typedef __attribute__((ext_vector_type(16))) _Float16 v16h;
typedef __attribute__((ext_vector_type(8)))  _Float16 v8h;
typedef __attribute__((ext_vector_type(8)))  float    v8f;

typedef __attribute__((ext_vector_type(4))) unsigned tdm_u4;
typedef __attribute__((ext_vector_type(8))) int      tdm_i8;
typedef __attribute__((ext_vector_type(4))) int      tdm_i4;

#define NPTS   12288
#define INDIM  256
#define DQK    64
#define KPAD   288     // 259 padded to 9 * 32
#define NKQV   384     // 64 + 64 + 256 fused QKV output columns
#define NCHUNK (NPTS / 32)
#define CLAMPV 100.0f

// ---------------------------------------------------------------------------
// Fragment helpers (CDNA5 wave32 WMMA 16-bit layouts, ISA 7.12.2).
// A/B 16-bit 16x32 fragment per lane = two contiguous 16B chunks at
// K-offsets 8h and 16+8h (h = lane>>4).
// ---------------------------------------------------------------------------
__device__ __forceinline__ v16h load_frag(const _Float16* p, int h) {
    v8h lo = *(const v8h*)(p + 8 * h);
    v8h hi = *(const v8h*)(p + 16 + 8 * h);
    v16h a;
#pragma unroll
    for (int i = 0; i < 8; ++i) { a[i] = lo[i]; a[8 + i] = hi[i]; }
    return a;
}

__device__ __forceinline__ v8f wmma_f16(v16h a, v16h b, v8f c) {
    return __builtin_amdgcn_wmma_f32_16x16x32_f16(
        false, a, false, b, (short)0, c, false, false);
}

// ---------------------------------------------------------------------------
// TDM helper: issue a 2D tile load (gfx1250 tensor_load_to_lds).
// Descriptor bit-packing follows CDNA5 ISA.
// data_size = 8 bytes per element unit.
// This toolchain uses the 6-arg builtin:
//   (uint32x4 g0, int32x8 g1, int32x4 g2, int32x4 g3, int32x8 pad, i32 cpol)
// ---------------------------------------------------------------------------
#if __has_builtin(__builtin_amdgcn_tensor_load_to_lds)
#define HAVE_TDM 1
__device__ __forceinline__ void tdm_load_2d(unsigned lds_off,
                                            unsigned long long gaddr,
                                            unsigned tile_d0_units,
                                            unsigned tile_d1,
                                            unsigned long long stride_units) {
    const unsigned TD = 1u << 20;               // huge tensor dims: no OOB
    tdm_u4 g0;
    g0[0] = 1u;                                  // count=1, user descriptor
    g0[1] = lds_off;                             // lds_addr (bytes)
    g0[2] = (unsigned)(gaddr & 0xffffffffu);     // global_addr[31:0]
    g0[3] = (unsigned)((gaddr >> 32) & 0x01ffffffu) | (2u << 30); // type=2
    tdm_i8 g1;
    g1[0] = (int)(3u << 16);                     // wg_mask=0, data_size=8B
    g1[1] = (int)((TD & 0xffffu) << 16);         // tensor_dim0[15:0]
    g1[2] = (int)((TD >> 16) | ((TD & 0xffffu) << 16));   // td0 hi | td1 lo
    g1[3] = (int)((TD >> 16) | (tile_d0_units << 16));    // td1 hi | tile_dim0
    g1[4] = (int)(tile_d1 & 0xffffu);            // tile_dim1 (0 = 1D)
    g1[5] = (int)(unsigned)(stride_units & 0xffffffffu);  // dim0 stride lo
    g1[6] = (int)(unsigned)((stride_units >> 32) & 0xffffu);
    g1[7] = 0;
    tdm_i4 gz4 = {0, 0, 0, 0};
    tdm_i8 gz8 = {0, 0, 0, 0, 0, 0, 0, 0};
    __builtin_amdgcn_tensor_load_to_lds(g0, g1, gz4, gz4, gz8, 0);
}
#else
#define HAVE_TDM 0
#endif

__device__ __forceinline__ void wait_tensorcnt0() {
#if __has_builtin(__builtin_amdgcn_s_wait_tensorcnt)
    __builtin_amdgcn_s_wait_tensorcnt(0);
#else
    asm volatile("s_wait_tensorcnt 0x0" ::: "memory");
#endif
}

__device__ __forceinline__ unsigned lds_addr_of(const void* p) {
    return (unsigned)(unsigned long long)(p);    // generic addr low 32 = LDS off
}

// ---------------------------------------------------------------------------
// Kernel 1: L2-normalize features, clamp coords, pack padded f16 x [N, KPAD]
// ---------------------------------------------------------------------------
__global__ void k_pack_x(const float* __restrict__ feat,
                         const int* __restrict__ coords,
                         const float* __restrict__ ts,
                         _Float16* __restrict__ xh) {
    int row = blockIdx.x;
    int t   = threadIdx.x;                 // 256 threads
    __shared__ float red[8];
    float v  = feat[row * INDIM + t];
    float ss = v * v;
#pragma unroll
    for (int m = 16; m >= 1; m >>= 1) ss += __shfl_xor(ss, m, 32);
    if ((t & 31) == 0) red[t >> 5] = ss;
    __syncthreads();
    float tot = red[0] + red[1] + red[2] + red[3] +
                red[4] + red[5] + red[6] + red[7];
    float inv = 1.0f / (sqrtf(tot) + 1e-6f);
    xh[row * KPAD + t] = (_Float16)(v * inv);
    if (t < KPAD - INDIM) {                // cols 256..287: coords + zero pad
        float val = 0.0f;
        if (t < 3) {
            float tsv = fmaxf(ts[t], 1e-6f);
            float c   = (float)coords[row * 3 + t] / tsv;
            val = fminf(fmaxf(c, -CLAMPV), CLAMPV);
        }
        xh[row * KPAD + INDIM + t] = (_Float16)val;
    }
}

// ---------------------------------------------------------------------------
// Kernel 2: pack [Wq|Wk|Wv] K-major:  WhT[col][k] (zero-padded to KPAD)
// ---------------------------------------------------------------------------
__global__ void k_pack_w(const float* __restrict__ Wq,
                         const float* __restrict__ Wk,
                         const float* __restrict__ Wv,
                         _Float16* __restrict__ WhT) {
    int idx = blockIdx.x * blockDim.x + threadIdx.x;
    if (idx >= NKQV * KPAD) return;
    int col = idx / KPAD;
    int k   = idx % KPAD;
    float val = 0.0f;
    if (k < INDIM + 3) {
        if      (col < 64)  val = Wq[k * DQK   + col];
        else if (col < 128) val = Wk[k * DQK   + (col - 64)];
        else                val = Wv[k * INDIM + (col - 128)];
    }
    WhT[col * KPAD + k] = (_Float16)val;
}

// ---------------------------------------------------------------------------
// Kernel 3: fused QKV GEMM.  One wave per 16x16 tile, 9 WMMA K-steps.
// Q, K row-major f16 [N,64];  V transposed f16 [256, N].
// ---------------------------------------------------------------------------
__global__ __launch_bounds__(256) void k_qkv(
        const _Float16* __restrict__ xh, const _Float16* __restrict__ WhT,
        const float* __restrict__ bq, const float* __restrict__ bk,
        const float* __restrict__ bv,
        _Float16* __restrict__ Qh, _Float16* __restrict__ Kh,
        _Float16* __restrict__ VhT) {
    const int NT = NKQV / 16;                   // 24 column tiles
    int wave = threadIdx.x >> 5;
    int lane = threadIdx.x & 31;
    int tile = blockIdx.x * 8 + wave;
    int tm = tile / NT, tn = tile % NT;
    int n = lane & 15, h = lane >> 4;

    const _Float16* arow = xh  + (tm * 16 + n) * KPAD;
    const _Float16* brow = WhT + (tn * 16 + n) * KPAD;
    v8f c = {};
    for (int kb = 0; kb < KPAD / 32; ++kb) {
        v16h a = load_frag(arow + kb * 32, h);
        v16h b = load_frag(brow + kb * 32, h);
        c = wmma_f16(a, b, c);
    }
    int col = tn * 16 + n;
    float bias = (col < 64) ? bq[col]
               : (col < 128) ? bk[col - 64] : bv[col - 128];
#pragma unroll
    for (int r = 0; r < 8; ++r) {
        int row = tm * 16 + r + 8 * h;          // C layout: VGPR r -> row r+8h
        _Float16 hv = (_Float16)(c[r] + bias);
        if      (col < 64)  Qh[row * DQK + col] = hv;
        else if (col < 128) Kh[row * DQK + (col - 64)] = hv;
        else                VhT[(size_t)(col - 128) * NPTS + row] = hv;
    }
}

// ---------------------------------------------------------------------------
// Kernel 4: flash attention, block-level K/V staging in LDS.
// 8 waves/block, one 16-query tile per wave; 32-key chunks double-buffered
// via TDM (tensor_load_to_lds) issued by wave 0, overlapped with WMMA.
// ---------------------------------------------------------------------------
__global__ __launch_bounds__(256) void k_attn(
        const _Float16* __restrict__ Qh, const _Float16* __restrict__ Kh,
        const _Float16* __restrict__ VhT,
        const float* __restrict__ feat, float* __restrict__ out) {
    __shared__ __align__(16) _Float16 Kbuf[2][32][64];    //  8 KB
    __shared__ __align__(16) _Float16 Vbuf[2][256][32];   // 32 KB (col-major V)
    __shared__ __align__(16) _Float16 ldsP[8][16][32];    //  8 KB

    int tid  = threadIdx.x;
    int wave = tid >> 5;
    int lane = tid & 31;
    int n = lane & 15, h = lane >> 4;
    int qbase = (blockIdx.x * 8 + wave) * 16;

    // Q fragments (dqk = 64 -> two 32-wide K-steps), loaded once
    v16h qf[2];
#pragma unroll
    for (int ks = 0; ks < 2; ++ks)
        qf[ks] = load_frag(Qh + (qbase + n) * DQK + ks * 32, h);

    float m[8], l[8];
    v8f acc[16];
#pragma unroll
    for (int r = 0; r < 8; ++r) { m[r] = -3.0e38f; l[r] = 0.0f; }
#pragma unroll
    for (int t = 0; t < 16; ++t) {
#pragma unroll
        for (int r = 0; r < 8; ++r) acc[t][r] = 0.0f;
    }

#if HAVE_TDM
    // ---- stage chunk 0 via TDM (wave 0 only), then barrier ----
    if (wave == 0) {
        // K chunk: 4 KB contiguous (32 rows x 128B) as 1D tile of 512 x 8B
        tdm_load_2d(lds_addr_of(&Kbuf[0][0][0]),
                    (unsigned long long)(const void*)Kh, 512u, 0u, 512ull);
        // V chunk: 256 rows x 64B, row stride NPTS*2 bytes = 3072 x 8B units
        tdm_load_2d(lds_addr_of(&Vbuf[0][0][0]),
                    (unsigned long long)(const void*)VhT, 8u, 256u, 3072ull);
        wait_tensorcnt0();
    }
    __syncthreads();
#else
    // ---- fallback: cooperative vector copy of chunk 0 ----
    {
        const v8h* ksrc = (const v8h*)Kh;
        ((v8h*)&Kbuf[0][0][0])[tid] = ksrc[tid];
#pragma unroll
        for (int j = 0; j < 4; ++j) {
            int e = j * 256 + tid, col = e >> 1, part = e & 1;
            *(v8h*)&Vbuf[0][col][part * 8] =
                *(const v8h*)(VhT + (size_t)col * NPTS + part * 8);
        }
    }
    __syncthreads();
#endif

    for (int kb = 0; kb < NCHUNK; ++kb) {
        int cur = kb & 1;
        int nxt = cur ^ 1;
        int keyn = (kb + 1) * 32;
        // ---- prefetch next chunk into the other buffer ----
        if (kb + 1 < NCHUNK) {
#if HAVE_TDM
            if (wave == 0) {
                tdm_load_2d(lds_addr_of(&Kbuf[nxt][0][0]),
                            (unsigned long long)(const void*)(Kh + keyn * DQK),
                            512u, 0u, 512ull);
                tdm_load_2d(lds_addr_of(&Vbuf[nxt][0][0]),
                            (unsigned long long)(const void*)(VhT + keyn),
                            8u, 256u, 3072ull);
            }
#else
            const v8h* ksrc = (const v8h*)(Kh + keyn * DQK);
            ((v8h*)&Kbuf[nxt][0][0])[tid] = ksrc[tid];
#pragma unroll
            for (int j = 0; j < 4; ++j) {
                int e = j * 256 + tid, col = e >> 1, part = e & 1;
                *(v8h*)&Vbuf[nxt][col][part * 8] =
                    *(const v8h*)(VhT + (size_t)col * NPTS + keyn + part * 8);
            }
#endif
        }

        // ---- S = Q @ K^T for two 16-key tiles (from LDS) ----
        v8f s0 = {}, s1 = {};
#pragma unroll
        for (int ks = 0; ks < 2; ++ks) {
            v16h b0 = load_frag(&Kbuf[cur][n][ks * 32], h);
            v16h b1 = load_frag(&Kbuf[cur][16 + n][ks * 32], h);
            s0 = wmma_f16(qf[ks], b0, s0);
            s1 = wmma_f16(qf[ks], b1, s1);
        }
        // ---- scale, clip, online softmax (per row r+8h) ----
        float alpha[8];
#pragma unroll
        for (int r = 0; r < 8; ++r) {
            float a0 = fminf(fmaxf(s0[r] * 0.125f, -CLAMPV), CLAMPV);
            float a1 = fminf(fmaxf(s1[r] * 0.125f, -CLAMPV), CLAMPV);
            float cm = fmaxf(a0, a1);
#pragma unroll
            for (int msk = 8; msk >= 1; msk >>= 1)
                cm = fmaxf(cm, __shfl_xor(cm, msk, 16));
            float mn = fmaxf(m[r], cm);
            float al = __expf(m[r] - mn);
            float p0 = __expf(a0 - mn);
            float p1 = __expf(a1 - mn);
            float rs = p0 + p1;
#pragma unroll
            for (int msk = 8; msk >= 1; msk >>= 1)
                rs += __shfl_xor(rs, msk, 16);
            l[r] = l[r] * al + rs;
            m[r] = mn;
            alpha[r] = al;
            ldsP[wave][r + 8 * h][n]      = (_Float16)p0;
            ldsP[wave][r + 8 * h][16 + n] = (_Float16)p1;
        }
#pragma unroll
        for (int t = 0; t < 16; ++t) {
#pragma unroll
            for (int r = 0; r < 8; ++r) acc[t][r] *= alpha[r];
        }
        // P tile DS stores must land before the cross-lane fragment reload
        asm volatile("s_wait_dscnt 0x0" ::: "memory");
        v16h pf = load_frag(&ldsP[wave][n][0], h);
        // ---- acc += P @ V  (16 output column tiles, V from LDS) ----
#pragma unroll
        for (int t = 0; t < 16; ++t) {
            v16h bv_ = load_frag(&Vbuf[cur][t * 16 + n][0], h);
            acc[t] = wmma_f16(pf, bv_, acc[t]);
        }

        // ---- make next chunk visible to all waves ----
#if HAVE_TDM
        if (wave == 0) wait_tensorcnt0();
#endif
        __syncthreads();
    }

    // ---- epilogue: divide by (l + 1e-6), add residual ----
#pragma unroll
    for (int t = 0; t < 16; ++t) {
#pragma unroll
        for (int r = 0; r < 8; ++r) {
            int row = qbase + r + 8 * h;
            int col = t * 16 + n;
            out[row * INDIM + col] =
                acc[t][r] / (l[r] + 1e-6f) + feat[row * INDIM + col];
        }
    }
}

// ---------------------------------------------------------------------------
extern "C" void kernel_launch(void* const* d_in, const int* in_sizes, int n_in,
                              void* d_out, int out_size, void* d_ws, size_t ws_size,
                              hipStream_t stream) {
    const float* feat   = (const float*)d_in[0];
    const int*   coords = (const int*)  d_in[1];
    const float* ts     = (const float*)d_in[2];
    const float* Wq     = (const float*)d_in[3];
    const float* bq     = (const float*)d_in[4];
    const float* Wk     = (const float*)d_in[5];
    const float* bk     = (const float*)d_in[6];
    const float* Wv     = (const float*)d_in[7];
    const float* bv     = (const float*)d_in[8];
    float* out = (float*)d_out;

    char* ws = (char*)d_ws;
    _Float16* xh  = (_Float16*)(ws + 0);                 //  7,077,888 B
    _Float16* WhT = (_Float16*)(ws + 7077888);           //    221,184 B
    _Float16* Qh  = (_Float16*)(ws + 7299072);           //  1,572,864 B
    _Float16* Kh  = (_Float16*)(ws + 8871936);           //  1,572,864 B
    _Float16* VhT = (_Float16*)(ws + 10444800);          //  6,291,456 B

    k_pack_x<<<NPTS, 256, 0, stream>>>(feat, coords, ts, xh);
    k_pack_w<<<(NKQV * KPAD + 255) / 256, 256, 0, stream>>>(Wq, Wk, Wv, WhT);
    k_qkv<<<(NPTS / 16) * (NKQV / 16) / 8, 256, 0, stream>>>(
        xh, WhT, bq, bk, bv, Qh, Kh, VhT);
    k_attn<<<(NPTS / 16) / 8, 256, 0, stream>>>(Qh, Kh, VhT, feat, out);
}
